// GCNEncoder_21431886807831
// MI455X (gfx1250) — matrix operand order repaired and checked by reading, not compiled, not run on hardware
//
#include <hip/hip_runtime.h>
#include <hip/hip_bf16.h>
#include <stdint.h>

// CDNA5 (gfx1250) wave32 WMMA types
typedef __attribute__((ext_vector_type(16))) _Float16 v16h;
typedef __attribute__((ext_vector_type(8)))  _Float16 v8h;
typedef __attribute__((ext_vector_type(8)))  float    v8f;

#define DIN  256
#define HID1 256
#define HID2 128

// ---------------- degree / norm kernels ----------------
__global__ void k_deg_init(float* deg, int n) {
    int i = blockIdx.x * blockDim.x + threadIdx.x;
    if (i < n) deg[i] = 1.0f;  // self loop contributes 1
}

__global__ void k_deg_accum(const int* __restrict__ dst, float* deg, int e) {
    int i = blockIdx.x * blockDim.x + threadIdx.x;
    if (i < e) atomicAdd(&deg[dst[i]], 1.0f);
}

__global__ void k_deg_rsqrt(float* deg, int n) {
    int i = blockIdx.x * blockDim.x + threadIdx.x;
    if (i < n) deg[i] = rsqrtf(deg[i]);  // deg >= 1 always (self loops)
}

// ---------------- W -> f16 transpose: Wt[col][k] = (f16) W[k][col] ----------------
__global__ void k_w_f16_transpose(const float* __restrict__ W, _Float16* __restrict__ Wt,
                                  int K, int H) {
    int tid = blockIdx.x * blockDim.x + threadIdx.x;
    if (tid >= K * H) return;
    int k = tid / H;
    int c = tid - k * H;
    Wt[(size_t)c * K + k] = (_Float16)W[tid];
}

// ---------------- WMMA GEMM: C[nrows,HOUT] = act(A[nrows,KDIM]) @ W[KDIM,HOUT] ----------------
// Block: 256 threads = 8 waves; tile = 128 rows x 64 cols.
// W panel (64 cols x KDIM f16, transposed) staged into LDS via
// global_load_async_to_lds_b128 (ASYNCcnt) and shared by all waves.
// Each wave: one 16x32 f16 A-fragment reused across 4 column tiles -> 4 WMMAs/K-step.
// K-loop fully unrolled (8 steps) so the scheduler hoists A-loads/prefetches
// across WMMAs and the prefetch guard folds away.
template<int KDIM, int HOUT, bool RELU_IN>
__global__ __launch_bounds__(256)
void k_gemm_wmma(const float* __restrict__ A,
                 const _Float16* __restrict__ Wt,   // [HOUT][KDIM] f16 (col-major panels)
                 float* __restrict__ C,
                 int nrows) {
    constexpr int CBLK = 64;                // cols per block
    constexpr int LDK  = KDIM + 8;          // padded LDS row stride (f16 elems)
    __shared__ _Float16 sW[CBLK * LDK];

    const int tid  = threadIdx.x;
    const int row0 = blockIdx.x * 128;
    const int col0 = blockIdx.y * CBLK;

    // ---- stage W panel into LDS (async copy, b128 chunks) ----
    constexpr int CPR   = (KDIM * 2) / 16;  // b128 chunks per column row
    constexpr int TOTAL = CBLK * CPR;       // 2048 chunks for KDIM=256
#pragma unroll
    for (int c = tid; c < TOTAL; c += 256) {
        const int r   = c / CPR;
        const int off = (c - r * CPR) * 16;
        uint32_t lds_b = (uint32_t)(uintptr_t)(&sW[r * LDK]) + (uint32_t)off;
        uint64_t gaddr = (uint64_t)(uintptr_t)((const char*)(Wt + (size_t)(col0 + r) * KDIM) + off);
        asm volatile("global_load_async_to_lds_b128 %0, %1, off"
                     :: "v"(lds_b), "v"(gaddr) : "memory");
    }
    asm volatile("s_wait_asynccnt 0x0" ::: "memory");
    __syncthreads();

    const int lane = tid & 31;
    const int wv   = tid >> 5;             // wave id 0..7 -> row sub-tile
    const int m    = lane & 15;            // row (A) / col (B,C) within 16x16 tile
    const int kh   = (lane >> 4) << 3;     // A frag K sub-offset (ISA 16-bit A 16x32 layout)
    const int kb   = (lane >> 4) << 3;     // B frag K sub-offset (same K order as A)

    const int arow = min(row0 + wv * 16 + m, nrows - 1);
    const float* __restrict__ Arow = A + (size_t)arow * KDIM;

    v8f acc[4] = {};

#pragma unroll
    for (int k0 = 0; k0 < KDIM; k0 += 32) {
        if (k0 + 32 < KDIM)
            __builtin_prefetch(Arow + k0 + 32, 0, 1);  // global_prefetch_b8 (guard folds away)

        // ---- A fragment: elems 0..7 -> K=k0+kh+j ; 8..15 -> K=k0+16+kh+j
        float4 a0 = *(const float4*)(Arow + k0 + kh);
        float4 a1 = *(const float4*)(Arow + k0 + kh + 4);
        float4 a2 = *(const float4*)(Arow + k0 + 16 + kh);
        float4 a3 = *(const float4*)(Arow + k0 + 16 + kh + 4);
        if (RELU_IN) {
            a0.x = fmaxf(a0.x, 0.f); a0.y = fmaxf(a0.y, 0.f); a0.z = fmaxf(a0.z, 0.f); a0.w = fmaxf(a0.w, 0.f);
            a1.x = fmaxf(a1.x, 0.f); a1.y = fmaxf(a1.y, 0.f); a1.z = fmaxf(a1.z, 0.f); a1.w = fmaxf(a1.w, 0.f);
            a2.x = fmaxf(a2.x, 0.f); a2.y = fmaxf(a2.y, 0.f); a2.z = fmaxf(a2.z, 0.f); a2.w = fmaxf(a2.w, 0.f);
            a3.x = fmaxf(a3.x, 0.f); a3.y = fmaxf(a3.y, 0.f); a3.z = fmaxf(a3.z, 0.f); a3.w = fmaxf(a3.w, 0.f);
        }
        v16h a;
        a[0]  = (_Float16)a0.x; a[1]  = (_Float16)a0.y; a[2]  = (_Float16)a0.z; a[3]  = (_Float16)a0.w;
        a[4]  = (_Float16)a1.x; a[5]  = (_Float16)a1.y; a[6]  = (_Float16)a1.z; a[7]  = (_Float16)a1.w;
        a[8]  = (_Float16)a2.x; a[9]  = (_Float16)a2.y; a[10] = (_Float16)a2.z; a[11] = (_Float16)a2.w;
        a[12] = (_Float16)a3.x; a[13] = (_Float16)a3.y; a[14] = (_Float16)a3.z; a[15] = (_Float16)a3.w;

        // ---- 4 column tiles share the A fragment
#pragma unroll
        for (int t = 0; t < 4; ++t) {
            const _Float16* p = &sW[(t * 16 + m) * LDK + k0 + kb];  // K=k0+kb+j and k0+16+kb+j
            v8h lo = *(const v8h*)p;          // ds_load_b128
            v8h hi = *(const v8h*)(p + 16);   // ds_load_b128
            v16h b = __builtin_shufflevector(lo, hi,
                                             0, 1, 2, 3, 4, 5, 6, 7,
                                             8, 9, 10, 11, 12, 13, 14, 15);
            acc[t] = __builtin_amdgcn_wmma_f32_16x16x32_f16(
                false, a, false, b, (short)0, acc[t], false, false);
        }
    }

    // ---- store: lanes 0-15: vgpr i -> row base+i ; lanes 16-31: row base+8+i
    const int rbase = row0 + wv * 16 + ((lane >> 4) << 3);
    if (row0 + 128 <= nrows) {
        // fast path: full block in range, no per-store guards
#pragma unroll
        for (int t = 0; t < 4; ++t) {
            const int col = col0 + t * 16 + m;
#pragma unroll
            for (int i = 0; i < 8; ++i)
                C[(size_t)(rbase + i) * HOUT + col] = acc[t][i];
        }
    } else {
        // tail block only (1 of ~782)
#pragma unroll
        for (int t = 0; t < 4; ++t) {
            const int col = col0 + t * 16 + m;
#pragma unroll
            for (int i = 0; i < 8; ++i) {
                int r = rbase + i;
                if (r < nrows) C[(size_t)r * HOUT + col] = acc[t][i];
            }
        }
    }
}

// ---------------- aggregation: agg[i,:] = bias + dinv[i]^2 * H[i,:] (self loop) ----------------
template<int H>
__global__ void k_agg_init(const float* __restrict__ Hbuf,
                           const float* __restrict__ bias,
                           const float* __restrict__ dinv,
                           float* __restrict__ agg, int n) {
    int tid = blockIdx.x * blockDim.x + threadIdx.x;
    if (tid >= n * H) return;
    int row = tid / H;
    int col = tid - row * H;
    float di = dinv[row];
    agg[tid] = bias[col] + di * di * Hbuf[tid];
}

// ---------------- edge scatter: agg[dst,:] += dinv[s]*dinv[d] * H[src,:] ----------------
template<int H>  // H/4 lanes per edge: float4 gather + 4 f32 atomics per lane
__global__ void k_agg_edges(const int* __restrict__ src,
                            const int* __restrict__ dst,
                            const float* __restrict__ dinv,
                            const float* __restrict__ Hbuf,
                            float* __restrict__ agg, int e) {
    constexpr int TPE = H / 4;
    int t    = blockIdx.x * blockDim.x + threadIdx.x;
    int edge = t / TPE;
    int part = t - edge * TPE;
    if (edge >= e) return;
    int s = src[edge], d = dst[edge];
    float w = dinv[s] * dinv[d];
    const float4 v = *(const float4*)(Hbuf + (size_t)s * H + part * 4);
    float* out = agg + (size_t)d * H + part * 4;
    atomicAdd(out + 0, w * v.x);
    atomicAdd(out + 1, w * v.y);
    atomicAdd(out + 2, w * v.z);
    atomicAdd(out + 3, w * v.w);
}

extern "C" void kernel_launch(void* const* d_in, const int* in_sizes, int n_in,
                              void* d_out, int out_size, void* d_ws, size_t ws_size,
                              hipStream_t stream) {
    const float* x  = (const float*)d_in[0];
    const float* W1 = (const float*)d_in[1];
    const float* b1 = (const float*)d_in[2];
    const float* W2 = (const float*)d_in[3];
    const float* b2 = (const float*)d_in[4];
    const int*   ei = (const int*)d_in[5];

    const int N = in_sizes[0] / DIN;
    const int E = in_sizes[5] / 2;
    const int* src = ei;
    const int* dst = ei + E;

    // ---- workspace carve: Wt1 | Wt2 | dinv | bufA | bufB ----
    char* ws = (char*)d_ws;
    size_t o = 0;
    _Float16* Wt1 = (_Float16*)(ws + o); o = (o + (size_t)DIN  * HID1 * 2 + 255) & ~(size_t)255;
    _Float16* Wt2 = (_Float16*)(ws + o); o = (o + (size_t)HID1 * HID2 * 2 + 255) & ~(size_t)255;
    float*   dinv = (float*)(ws + o);    o = (o + (size_t)N * 4 + 255) & ~(size_t)255;
    float*   bufA = (float*)(ws + o);    o = (o + (size_t)N * HID1 * 4 + 255) & ~(size_t)255;
    float*   bufB = (float*)(ws + o);
    float*   out  = (float*)d_out;

    const int T = 256;

    // weights -> f16 transposed (tiny, once per launch)
    k_w_f16_transpose<<<(DIN * HID1 + T - 1) / T, T, 0, stream>>>(W1, Wt1, DIN, HID1);
    k_w_f16_transpose<<<(HID1 * HID2 + T - 1) / T, T, 0, stream>>>(W2, Wt2, HID1, HID2);

    // degrees -> dinv
    k_deg_init <<<(N + T - 1) / T, T, 0, stream>>>(dinv, N);
    k_deg_accum<<<(E + T - 1) / T, T, 0, stream>>>(dst, dinv, E);
    k_deg_rsqrt<<<(N + T - 1) / T, T, 0, stream>>>(dinv, N);

    // ---- layer 1: H = x @ W1 ; agg1 = scatter(norm * H) + b1 ----
    dim3 g1((N + 127) / 128, HID1 / 64);
    k_gemm_wmma<DIN, HID1, false><<<g1, 256, 0, stream>>>(x, Wt1, bufA, N);

    long long tot1 = (long long)N * HID1;
    k_agg_init<HID1><<<(unsigned)((tot1 + T - 1) / T), T, 0, stream>>>(bufA, b1, dinv, bufB, N);
    long long th1 = (long long)E * (HID1 / 4);
    k_agg_edges<HID1><<<(unsigned)((th1 + T - 1) / T), T, 0, stream>>>(src, dst, dinv, bufA, bufB, E);

    // ---- layer 2: H2 = relu(agg1) @ W2 (ReLU fused into A-load); agg2 -> d_out ----
    dim3 g2((N + 127) / 128, HID2 / 64);
    k_gemm_wmma<HID1, HID2, true><<<g2, 256, 0, stream>>>(bufB, Wt2, bufA, N);

    long long tot2 = (long long)N * HID2;
    k_agg_init<HID2><<<(unsigned)((tot2 + T - 1) / T), T, 0, stream>>>(bufA, b2, dinv, out, N);
    long long th2 = (long long)E * (HID2 / 4);
    k_agg_edges<HID2><<<(unsigned)((th2 + T - 1) / T), T, 0, stream>>>(src, dst, dinv, bufA, out, E);
}
